// GPT_5093831213187
// MI455X (gfx1250) — matrix operand order repaired and checked by reading, not compiled, not run on hardware
//
#include <hip/hip_runtime.h>
#include <hip/hip_bf16.h>
#include <math.h>
#include <stdint.h>

// ---------------- model dims ----------------
#define HDIM   768
#define NHEAD  12
#define DHEAD  64
#define LAYERS 12
#define SEQ    2048
#define BATCH  2
#define MTOK   (BATCH * SEQ)
#define VOCAB  50257

typedef __bf16 bf16x16 __attribute__((ext_vector_type(16)));
typedef float  floatx8 __attribute__((ext_vector_type(8)));

union Frag { bf16x16 v; uint4 q[2]; };

static __device__ __forceinline__ unsigned short f2bf(float f) {
    unsigned int u = __float_as_uint(f);
    u += 0x7fffu + ((u >> 16) & 1u);
    return (unsigned short)(u >> 16);
}
static __device__ __forceinline__ unsigned int pack2bf(float lo, float hi) {
    return (unsigned int)f2bf(lo) | ((unsigned int)f2bf(hi) << 16);
}

// generic LDS pointer -> 32-bit allocation-relative offset (addr[31:0])
static __device__ __forceinline__ unsigned ldsOff(const void* p) {
    return (unsigned)(uintptr_t)p;
}
// one 16B async global->LDS copy (tracked by ASYNCcnt)
static __device__ __forceinline__ void asyncCopy16(unsigned lds, const void* g) {
    asm volatile("global_load_async_to_lds_b128 %0, %1, off"
                 :: "v"(lds), "v"((unsigned long long)(uintptr_t)g) : "memory");
}
// two consecutive 16B copies (offset applies to both LDS and global address)
static __device__ __forceinline__ void asyncCopy32(unsigned lds, const void* g) {
    asm volatile("global_load_async_to_lds_b128 %0, %1, off\n\t"
                 "global_load_async_to_lds_b128 %0, %1, off offset:16"
                 :: "v"(lds), "v"((unsigned long long)(uintptr_t)g) : "memory");
}
static __device__ __forceinline__ void asyncCopy64(unsigned lds, const void* g) {
    asm volatile("global_load_async_to_lds_b128 %0, %1, off\n\t"
                 "global_load_async_to_lds_b128 %0, %1, off offset:16\n\t"
                 "global_load_async_to_lds_b128 %0, %1, off offset:32\n\t"
                 "global_load_async_to_lds_b128 %0, %1, off offset:48"
                 :: "v"(lds), "v"((unsigned long long)(uintptr_t)g) : "memory");
}
static __device__ __forceinline__ void waitAsync0() {
    asm volatile("s_wait_asynccnt 0x0" ::: "memory");
}

// 16-lane xor-swap via ds_swizzle: reduces within each half-wave, matching the
// WMMA C layout (each 16-lane half owns 8 full rows).
#define SWZ(x, m) __uint_as_float((unsigned)__builtin_amdgcn_ds_swizzle( \
        (int)__float_as_uint(x), 0x1f | ((m) << 10)))
static __device__ __forceinline__ float rowmax16(float x) {
    x = fmaxf(x, SWZ(x, 1)); x = fmaxf(x, SWZ(x, 2));
    x = fmaxf(x, SWZ(x, 4)); x = fmaxf(x, SWZ(x, 8));
    return x;
}
static __device__ __forceinline__ float rowsum16(float x) {
    x += SWZ(x, 1); x += SWZ(x, 2); x += SWZ(x, 4); x += SWZ(x, 8);
    return x;
}

// ---------------- embedding ----------------
__global__ void gpt_embed_kernel(const int* __restrict__ ids,
                                 const float* __restrict__ wte,
                                 const float* __restrict__ wpe,
                                 float* __restrict__ x) {
    int idx = blockIdx.x * 256 + threadIdx.x;
    int m = idx / HDIM, c = idx % HDIM;
    int s = m % SEQ;
    int tok = ids[m];
    x[idx] = wte[(size_t)tok * HDIM + c] + wpe[(size_t)s * HDIM + c];
}

// ---------------- layernorm over H=768 ----------------
__global__ void gpt_ln_kernel(const float* __restrict__ x,
                              const float* __restrict__ g,
                              const float* __restrict__ b,
                              unsigned short* __restrict__ outB,
                              float* __restrict__ outF) {
    __shared__ float red[256];
    const int m = blockIdx.x;
    const int t = threadIdx.x;
    const float* row = x + (size_t)m * HDIM;
    float v[3];
    float s = 0.f;
    for (int i = 0; i < 3; ++i) { v[i] = row[t + 256 * i]; s += v[i]; }
    red[t] = s; __syncthreads();
    for (int off = 128; off > 0; off >>= 1) {
        if (t < off) red[t] += red[t + off];
        __syncthreads();
    }
    float mean = red[0] * (1.0f / HDIM);
    __syncthreads();
    float s2 = 0.f;
    for (int i = 0; i < 3; ++i) { float d = v[i] - mean; s2 += d * d; }
    red[t] = s2; __syncthreads();
    for (int off = 128; off > 0; off >>= 1) {
        if (t < off) red[t] += red[t + off];
        __syncthreads();
    }
    float rstd = rsqrtf(red[0] * (1.0f / HDIM) + 1e-12f);
    for (int i = 0; i < 3; ++i) {
        int c = t + 256 * i;
        float y = (v[i] - mean) * rstd * g[c] + b[c];
        if (outB) outB[(size_t)m * HDIM + c] = f2bf(y);
        if (outF) outF[(size_t)m * HDIM + c] = y;
    }
}

// ---------------- WMMA bf16 GEMM, double-buffered, async A staging ----------------
// 256 thr = 8 waves; block tile 128(M) x 64(N); K step 32; 4 acc per wave.
template <bool WT, bool BOUNDED>
__global__ void __launch_bounds__(256)
gpt_gemm_bf16_wmma(const unsigned short* __restrict__ A,  // M x K bf16
                   const float* __restrict__ W,           // K x N (N x K if WT)
                   const float* __restrict__ bias,
                   const float* __restrict__ resid,       // M x N
                   float* __restrict__ outF,              // ld = ldOut
                   unsigned short* __restrict__ outB,     // ld = N
                   int N, int K, int Nreal, int Mstore, int ldOut, int doGelu) {
    __shared__ __align__(16) unsigned short sA[2][128 * 32];
    __shared__ __align__(16) unsigned short sB[2][64 * 32];

    const int tid  = threadIdx.x;
    const int wave = tid >> 5;
    const int lane = tid & 31;
    const int half = lane >> 4;
    const int lr   = lane & 15;
    const int m0 = blockIdx.y * 128;
    const int n0 = blockIdx.x * 64;

    // staging roles
    const int rowA = tid >> 1, segA = tid & 1;
    const int nB = tid & 63, kkB = (tid >> 6) * 8;
    const bool nOK = !BOUNDED || (n0 + nB < Nreal);

    // A tile: 32B/thread, async DMA straight into the LDS double buffer
    auto asyncLoadA = [&](int k0, int buf) {
        asyncCopy32(ldsOff(&sA[buf][rowA * 32 + segA * 16]),
                    A + (size_t)(m0 + rowA) * K + k0 + segA * 16);
    };
    // W tile: fp32 loads -> bf16 pack in registers (needs VALU convert)
    float rw[8];
    auto loadW = [&](int k0) {
#pragma unroll
        for (int i = 0; i < 8; ++i) {
            float w = 0.0f;
            if (nOK)
                w = WT ? W[(size_t)(n0 + nB) * K + (k0 + kkB + i)]
                       : W[(size_t)(k0 + kkB + i) * N + (n0 + nB)];
            rw[i] = w;
        }
    };
    auto storeW = [&](int buf) {
        uint4 w;
        w.x = pack2bf(rw[0], rw[1]);
        w.y = pack2bf(rw[2], rw[3]);
        w.z = pack2bf(rw[4], rw[5]);
        w.w = pack2bf(rw[6], rw[7]);
        *(uint4*)&sB[buf][nB * 32 + kkB] = w;
    };

    const floatx8 vzero = {0.f, 0.f, 0.f, 0.f, 0.f, 0.f, 0.f, 0.f};
    floatx8 acc[4];
#pragma unroll
    for (int j = 0; j < 4; ++j) acc[j] = vzero;

    asyncLoadA(0, 0);
    loadW(0);
    int buf = 0;
    for (int k0 = 0; k0 < K; k0 += 32) {
        storeW(buf);
        waitAsync0();       // this wave's A-tile DMA into sA[buf] complete
        __syncthreads();    // publish sA[buf]/sB[buf] block-wide
        if (k0 + 32 < K) {  // overlap next tile (async DMA + W loads) with WMMAs
            asyncLoadA(k0 + 32, buf ^ 1);
            loadW(k0 + 32);
        }

        Frag a;
        {
            int row = wave * 16 + lr;
            a.q[0] = *(const uint4*)&sA[buf][row * 32 + half * 8];
            a.q[1] = *(const uint4*)&sA[buf][row * 32 + 16 + half * 8];
        }
#pragma unroll
        for (int j = 0; j < 4; ++j) {
            Frag bf;
            int col = j * 16 + lr;
            bf.q[0] = *(const uint4*)&sB[buf][col * 32 + half * 8];
            bf.q[1] = *(const uint4*)&sB[buf][col * 32 + 16 + half * 8];
            acc[j] = __builtin_amdgcn_wmma_f32_16x16x32_bf16(
                false, a.v, false, bf.v, (short)0, acc[j], false, false);
        }
        buf ^= 1;
    }

    // epilogue
#pragma unroll
    for (int j = 0; j < 4; ++j) {
        int n = n0 + j * 16 + lr;
        if (BOUNDED && n >= Nreal) continue;
        float bv = bias ? bias[n] : 0.0f;
#pragma unroll
        for (int r = 0; r < 8; ++r) {
            int mg = m0 + wave * 16 + half * 8 + r;
            if (BOUNDED && mg >= Mstore) continue;
            float v = acc[j][r] + bv;
            if (doGelu) v = 0.5f * v * (1.0f + erff(v * 0.70710678118654752f));
            if (resid) v += resid[(size_t)mg * N + n];
            if (outF) outF[(size_t)mg * ldOut + n] = v;
            if (outB) outB[(size_t)mg * N + n] = f2bf(v);
        }
    }
}

// ---------------- WMMA flash attention ----------------
// block = (128-query chunk, batch*head); 8 waves x 16 queries; 32-key tiles.
__global__ void __launch_bounds__(256)
gpt_flash_attn_wmma(const unsigned short* __restrict__ qkvb,  // [MTOK][3H] bf16
                    unsigned short* __restrict__ out) {       // [MTOK][H] bf16
    __shared__ __align__(16) unsigned short qs[128 * 64];   // Q chunk
    __shared__ __align__(16) unsigned short ks[32 * 64];    // K tile [key][d]
    __shared__ __align__(16) unsigned short vs[64 * 32];    // V tile transposed [d][key]
    __shared__ __align__(16) unsigned short ps[8][16 * 32]; // per-wave P tile

    const int tid = threadIdx.x;
    const int wave = tid >> 5, lane = tid & 31;
    const int half = lane >> 4, lr = lane & 15;
    const int q0 = blockIdx.x * 128;
    const int bh = blockIdx.y;
    const int b = bh / NHEAD, h = bh % NHEAD;
    const int rs3 = 3 * HDIM;
    const unsigned short* qbase = qkvb + (size_t)(b * SEQ) * rs3 + h * DHEAD;
    const unsigned short* kbase = qbase + HDIM;
    const unsigned short* vbase = qbase + 2 * HDIM;

    // stage the 128x64 Q chunk: 64B/thread async DMA
    {
        int row = tid >> 1, seg = tid & 1;
        asyncCopy64(ldsOff(&qs[row * 64 + seg * 32]),
                    qbase + (size_t)(q0 + row) * rs3 + seg * 32);
    }
    waitAsync0();
    __syncthreads();

    Frag qf[2];
    {
        int row = wave * 16 + lr;
#pragma unroll
        for (int c = 0; c < 2; ++c) {
            qf[c].q[0] = *(const uint4*)&qs[row * 64 + c * 32 + half * 8];
            qf[c].q[1] = *(const uint4*)&qs[row * 64 + c * 32 + 16 + half * 8];
        }
    }

    const floatx8 vzero = {0.f, 0.f, 0.f, 0.f, 0.f, 0.f, 0.f, 0.f};
    floatx8 oacc[4];
#pragma unroll
    for (int j = 0; j < 4; ++j) oacc[j] = vzero;
    float mrow[8], lrow[8];
#pragma unroll
    for (int r = 0; r < 8; ++r) { mrow[r] = -1e30f; lrow[r] = 0.f; }

    const int wq_hi = q0 + wave * 16 + 15;
    const int ktiles = (q0 + 128) / 32;

    for (int it = 0; it < ktiles; ++it) {
        const int kt = it * 32;
        __syncthreads();  // previous tile fully consumed before restaging
        // K tile: 16B/thread async DMA (in flight while V is transposed below)
        {
            int key = tid >> 3, seg = tid & 7;
            asyncCopy16(ldsOff(&ks[key * 64 + seg * 8]),
                        kbase + (size_t)(kt + key) * rs3 + seg * 8);
        }
        // V tile transposed: vs[d][key], 8 keys/thread (coalesced across d)
        {
            int dd = tid & 63, kg = tid >> 6;
            uint4 w;
            unsigned int pk[4];
#pragma unroll
            for (int i = 0; i < 4; ++i) {
                unsigned short a0 = vbase[(size_t)(kt + kg * 8 + 2 * i) * rs3 + dd];
                unsigned short a1 = vbase[(size_t)(kt + kg * 8 + 2 * i + 1) * rs3 + dd];
                pk[i] = (unsigned int)a0 | ((unsigned int)a1 << 16);
            }
            w.x = pk[0]; w.y = pk[1]; w.z = pk[2]; w.w = pk[3];
            *(uint4*)&vs[dd * 32 + kg * 8] = w;
        }
        waitAsync0();
        __syncthreads();

        if (kt <= wq_hi) {  // wave-uniform: EXEC stays all-ones for WMMA
            floatx8 s[2];
            s[0] = vzero; s[1] = vzero;
#pragma unroll
            for (int st = 0; st < 2; ++st) {
#pragma unroll
                for (int c = 0; c < 2; ++c) {
                    Frag kf;
                    int key = st * 16 + lr;
                    kf.q[0] = *(const uint4*)&ks[key * 64 + c * 32 + half * 8];
                    kf.q[1] = *(const uint4*)&ks[key * 64 + c * 32 + 16 + half * 8];
                    s[st] = __builtin_amdgcn_wmma_f32_16x16x32_bf16(
                        false, qf[c].v, false, kf.v, (short)0, s[st], false, false);
                }
            }
            // causal mask + 1/sqrt(64) scale + online softmax
#pragma unroll
            for (int r = 0; r < 8; ++r) {
                int qrow = q0 + wave * 16 + half * 8 + r;
                float a0 = (kt + lr      <= qrow) ? s[0][r] * 0.125f : -INFINITY;
                float a1 = (kt + 16 + lr <= qrow) ? s[1][r] * 0.125f : -INFINITY;
                float mx = rowmax16(fmaxf(a0, a1));
                float mn = fmaxf(mrow[r], mx);
                float p0 = __expf(a0 - mn);
                float p1 = __expf(a1 - mn);
                float scale = __expf(mrow[r] - mn);
                lrow[r] = lrow[r] * scale + rowsum16(p0 + p1);
                mrow[r] = mn;
#pragma unroll
                for (int j = 0; j < 4; ++j) oacc[j][r] *= scale;
                int prow = half * 8 + r;
                ps[wave][prow * 32 + lr]      = f2bf(p0);
                ps[wave][prow * 32 + 16 + lr] = f2bf(p1);
            }
            // P x V (wave-local LDS: per-wave in-order, no barrier needed)
            Frag pf;
            pf.q[0] = *(const uint4*)&ps[wave][lr * 32 + half * 8];
            pf.q[1] = *(const uint4*)&ps[wave][lr * 32 + 16 + half * 8];
#pragma unroll
            for (int j = 0; j < 4; ++j) {
                Frag vf;
                int col = j * 16 + lr;
                vf.q[0] = *(const uint4*)&vs[col * 32 + half * 8];
                vf.q[1] = *(const uint4*)&vs[col * 32 + 16 + half * 8];
                oacc[j] = __builtin_amdgcn_wmma_f32_16x16x32_bf16(
                    false, pf.v, false, vf.v, (short)0, oacc[j], false, false);
            }
        }
    }

    // write O / l
#pragma unroll
    for (int j = 0; j < 4; ++j) {
#pragma unroll
        for (int r = 0; r < 8; ++r) {
            int q = q0 + wave * 16 + half * 8 + r;
            int d = j * 16 + lr;
            out[(size_t)(b * SEQ + q) * HDIM + h * DHEAD + d] = f2bf(oacc[j][r] / lrow[r]);
        }
    }
}

// ---------------- LM-head A prep: 2 live rows, zero-padded to 128 ----------------
__global__ void gpt_lm_prep_kernel(const float* __restrict__ xf,
                                   unsigned short* __restrict__ xpad) {
    int idx = blockIdx.x * 256 + threadIdx.x;
    int row = idx / HDIM, c = idx % HDIM;
    float v = 0.f;
    if (row == 0)      v = xf[(size_t)(SEQ - 1) * HDIM + c];
    else if (row == 1) v = xf[(size_t)(2 * SEQ - 1) * HDIM + c];
    xpad[idx] = f2bf(v);
}

// ---------------- host orchestration ----------------
extern "C" void kernel_launch(void* const* d_in, const int* in_sizes, int n_in,
                              void* d_out, int out_size, void* d_ws, size_t ws_size,
                              hipStream_t stream) {
    const int*   ids     = (const int*)d_in[0];
    const float* wte     = (const float*)d_in[1];
    const float* wpe     = (const float*)d_in[2];
    const float* ln1_g   = (const float*)d_in[3];
    const float* ln1_b   = (const float*)d_in[4];
    const float* qkv_w   = (const float*)d_in[5];
    const float* qkv_b   = (const float*)d_in[6];
    const float* dense_w = (const float*)d_in[7];
    const float* dense_b = (const float*)d_in[8];
    const float* ln2_g   = (const float*)d_in[9];
    const float* ln2_b   = (const float*)d_in[10];
    const float* fc1_w   = (const float*)d_in[11];
    const float* fc2_w   = (const float*)d_in[12];
    const float* fc2_b   = (const float*)d_in[13];
    const float* lnf_g   = (const float*)d_in[14];
    const float* lnf_b   = (const float*)d_in[15];

    char* ws = (char*)d_ws;
    size_t off = 0;
    float* x  = (float*)(ws + off);  off += (size_t)MTOK * HDIM * 4;
    float* xf = (float*)(ws + off);  off += (size_t)MTOK * HDIM * 4;
    unsigned short* h     = (unsigned short*)(ws + off); off += (size_t)MTOK * HDIM * 2;
    unsigned short* qkvb  = (unsigned short*)(ws + off); off += (size_t)MTOK * 3 * HDIM * 2;
    unsigned short* attnb = (unsigned short*)(ws + off); off += (size_t)MTOK * HDIM * 2;
    unsigned short* mlp1  = (unsigned short*)(ws + off); off += (size_t)MTOK * 2 * HDIM * 2;
    unsigned short* xpad  = (unsigned short*)(ws + off); off += (size_t)128 * HDIM * 2;
    (void)ws_size; (void)n_in; (void)in_sizes; (void)out_size;

    gpt_embed_kernel<<<(MTOK * HDIM) / 256, 256, 0, stream>>>(ids, wte, wpe, x);

    const dim3 gQKV(3 * HDIM / 64, MTOK / 128);
    const dim3 gPRJ(HDIM / 64, MTOK / 128);
    const dim3 gFC1(2 * HDIM / 64, MTOK / 128);
    const dim3 gATT(SEQ / 128, BATCH * NHEAD);

    for (int l = 0; l < LAYERS; ++l) {
        const float* g1 = ln1_g + (size_t)l * HDIM;
        const float* b1 = ln1_b + (size_t)l * HDIM;
        const float* qw = qkv_w + (size_t)l * HDIM * 3 * HDIM;
        const float* qb = qkv_b + (size_t)l * 3 * HDIM;
        const float* dw = dense_w + (size_t)l * HDIM * HDIM;
        const float* db = dense_b + (size_t)l * HDIM;
        const float* g2 = ln2_g + (size_t)l * HDIM;
        const float* b2 = ln2_b + (size_t)l * HDIM;
        const float* f1 = fc1_w + (size_t)l * HDIM * 2 * HDIM;
        const float* f2 = fc2_w + (size_t)l * 2 * HDIM * HDIM;
        const float* fb = fc2_b + (size_t)l * HDIM;

        gpt_ln_kernel<<<MTOK, 256, 0, stream>>>(x, g1, b1, h, nullptr);
        gpt_gemm_bf16_wmma<false, false><<<gQKV, 256, 0, stream>>>(
            h, qw, qb, nullptr, nullptr, qkvb,
            3 * HDIM, HDIM, 3 * HDIM, MTOK, 3 * HDIM, 0);
        gpt_flash_attn_wmma<<<gATT, 256, 0, stream>>>(qkvb, attnb);
        gpt_gemm_bf16_wmma<false, false><<<gPRJ, 256, 0, stream>>>(
            attnb, dw, db, x, x, nullptr,
            HDIM, HDIM, HDIM, MTOK, HDIM, 0);
        gpt_ln_kernel<<<MTOK, 256, 0, stream>>>(x, g2, b2, h, nullptr);
        gpt_gemm_bf16_wmma<false, false><<<gFC1, 256, 0, stream>>>(
            h, f1, nullptr, nullptr, nullptr, mlp1,
            2 * HDIM, HDIM, 2 * HDIM, MTOK, 2 * HDIM, 1);
        gpt_gemm_bf16_wmma<false, false><<<gPRJ, 256, 0, stream>>>(
            mlp1, f2, fb, x, x, nullptr,
            HDIM, HDIM, HDIM, MTOK, HDIM, 0);
    }

    gpt_ln_kernel<<<MTOK, 256, 0, stream>>>(x, lnf_g, lnf_b, nullptr, xf);
    gpt_lm_prep_kernel<<<(128 * HDIM) / 256, 256, 0, stream>>>(xf, xpad);
    const int NPAD = ((VOCAB + 63) / 64) * 64;
    dim3 gLM(NPAD / 64, 1);
    gpt_gemm_bf16_wmma<true, true><<<gLM, 256, 0, stream>>>(
        xpad, wte, nullptr, nullptr, (float*)d_out, nullptr,
        NPAD, HDIM, VOCAB, BATCH, VOCAB, 0);
}
